// GaussianTokenDownsampler_67053029425670
// MI455X (gfx1250) — compile-verified, matrix operand order checked by reading, not run on hardware
//
#include <hip/hip_runtime.h>
#include <hip/hip_bf16.h>
#include <math.h>

typedef __attribute__((ext_vector_type(16))) _Float16 v16h;
typedef __attribute__((ext_vector_type(8)))  float    v8f;

union FragH { v16h h; uint4 u[2]; };

__device__ __forceinline__ v8f wmma16(v16h a, v16h b, v8f c) {
  // D = A(16x32 f16) * B(32x16 f16) + C(16x16 f32)
  return __builtin_amdgcn_wmma_f32_16x16x32_f16(false, a, false, b, (short)0, c, false, false);
}

__device__ __forceinline__ float gelu_exact(float v) {
  return 0.5f * v * (1.0f + erff(v * 0.70710678118654752f));
}

// ---------------------------------------------------------------------------
// CDNA5 async global->LDS staging (ASYNCcnt path), with synchronous fallback
// Builtin signature (from probe): (v4i AS1* gsrc, v4i AS3* ldsdst, Ii, Ii)
// ---------------------------------------------------------------------------
#if defined(__gfx1250__) && \
    __has_builtin(__builtin_amdgcn_global_load_async_to_lds_b128) && \
    __has_builtin(__builtin_amdgcn_s_wait_asynccnt)
#define USE_ASYNC_LDS 1
#else
#define USE_ASYNC_LDS 0
#endif

#if USE_ASYNC_LDS
typedef int v4i_raw __attribute__((vector_size(16)));
typedef __attribute__((address_space(1))) v4i_raw gv4i;
typedef __attribute__((address_space(3))) v4i_raw lv4i;
#endif

__device__ __forceinline__ void stage_b128(const void* g, void* l) {
#if USE_ASYNC_LDS
  __builtin_amdgcn_global_load_async_to_lds_b128((gv4i*)g, (lv4i*)l, 0, 0);
#else
  *(uint4*)l = *(const uint4*)g;
#endif
}

__device__ __forceinline__ void stage_wait() {
#if USE_ASYNC_LDS
  __builtin_amdgcn_s_wait_asynccnt(0);
#endif
}

// ---------------------------------------------------------------------------
// utility: f32 -> f16 convert
// ---------------------------------------------------------------------------
__global__ __launch_bounds__(256) void k_cvt(const float* __restrict__ in,
                                             _Float16* __restrict__ out, int n) {
  int i = blockIdx.x * 256 + threadIdx.x;
  if (i < n) out[i] = (_Float16)in[i];
}

// ---------------------------------------------------------------------------
// patch gather: x (B,4096,384) -> xp (B,1024,1536) f16, k = (di*2+dj)*384 + c
// ---------------------------------------------------------------------------
__global__ __launch_bounds__(256) void k_gather_xp(const float* __restrict__ x,
                                                   _Float16* __restrict__ xp, int n) {
  int i = blockIdx.x * 256 + threadIdx.x;
  if (i >= n) return;
  int b = i / (1024 * 1536);
  int rem = i - b * (1024 * 1536);
  int s = rem / 1536;
  int k = rem - s * 1536;
  int q = k / 384, c = k - q * 384;
  int di = q >> 1, dj = q & 1;
  int ii = s >> 5, jj = s & 31;
  int pos = ((2 * ii + di) << 6) + (2 * jj + dj);
  xp[i] = (_Float16)x[((size_t)b * 4096 + pos) * 384 + c];
}

// conv_w (O=384, C=384, 2, 2) -> wp (384, 1536) f16 with same k ordering
__global__ __launch_bounds__(256) void k_gather_wp(const float* __restrict__ w,
                                                   _Float16* __restrict__ wp, int n) {
  int i = blockIdx.x * 256 + threadIdx.x;
  if (i >= n) return;
  int o = i / 1536;
  int k = i - o * 1536;
  int q = k / 384, c = k - q * 384;
  int di = q >> 1, dj = q & 1;
  wp[i] = (_Float16)w[((size_t)(o * 384 + c) * 2 + di) * 2 + dj];
}

// ---------------------------------------------------------------------------
// generic WMMA GEMM:  C[M,N] = A[M,K](f16) * W[N,K]^T(f16) + bias
// block = 256 thr = 8 waves; tile 128(M) x 64(N); K step 32.
// A and W tiles double-buffered in LDS, staged with async global->LDS copies
// overlapped with the WMMAs of the previous K step (one barrier / step).
// mode: 0 f32 store, 1 f32 residual store, 2 f16 store, 3 gelu->f16,
//       4 f16 transposed per-batch store: out16[(b*N+n)*vtT + t], m = b*vtT+t
// ---------------------------------------------------------------------------
__global__ __launch_bounds__(256) void k_gemm(
    const _Float16* __restrict__ A, const _Float16* __restrict__ W,
    const float* __restrict__ bias, int M, int N, int K, int mode,
    float* __restrict__ out32, const float* __restrict__ res,
    _Float16* __restrict__ out16, int vtT) {
  __shared__ _Float16 wtile[2][64][32];
  __shared__ _Float16 atile[2][128][32];
  const int lane = threadIdx.x & 31;
  const int wave = threadIdx.x >> 5;
  const int sel  = (lane & 16) ? 8 : 0;      // A/B fragment K-chunk select
  const int n0 = blockIdx.x * 64;
  const int m0 = blockIdx.y * 128 + wave * 16;

  // staging coordinates
  const int wr = threadIdx.x >> 2;           // W tile row 0..63
  const int wc = (threadIdx.x & 3) * 8;      // halves 0,8,16,24
  const int ar = threadIdx.x >> 1;           // A tile row 0..127
  const int ac = (threadIdx.x & 1) * 16;     // halves 0 or 16
  const _Float16* Wsrc = W + (size_t)(n0 + wr) * K + wc;
  const _Float16* Asrc = A + (size_t)(blockIdx.y * 128 + ar) * K + ac;

  v8f acc[4];
  for (int j = 0; j < 4; ++j)
    for (int r = 0; r < 8; ++r) acc[j][r] = 0.f;

  auto stage = [&](int kt) {
    int bsel = kt & 1;
    stage_b128(Wsrc + kt * 32, &wtile[bsel][wr][wc]);
    stage_b128(Asrc + kt * 32, &atile[bsel][ar][ac]);
    stage_b128(Asrc + kt * 32 + 8, &atile[bsel][ar][ac + 8]);
  };

  const int nk = K / 32;
  stage(0);
  for (int kt = 0; kt < nk; ++kt) {
    stage_wait();
    __syncthreads();                 // tile kt visible; tile kt-1 fully consumed
    if (kt + 1 < nk) stage(kt + 1);  // DMA next tile while we compute
    if (kt + 2 < nk) __builtin_prefetch(Wsrc + (kt + 2) * 32, 0, 3);
    const int bsel = kt & 1;
    FragH a;
    const _Float16* ap = &atile[bsel][wave * 16 + (lane & 15)][sel];
    a.u[0] = *(const uint4*)ap;
    a.u[1] = *(const uint4*)(ap + 16);
#pragma unroll
    for (int j = 0; j < 4; ++j) {
      FragH bfr;
      const _Float16* bp = &wtile[bsel][j * 16 + (lane & 15)][sel];
      bfr.u[0] = *(const uint4*)bp;
      bfr.u[1] = *(const uint4*)(bp + 16);
      acc[j] = wmma16(a.h, bfr.h, acc[j]);
    }
  }

  const int rowoff = (lane & 16) ? 8 : 0;
#pragma unroll
  for (int j = 0; j < 4; ++j) {
    int n = n0 + j * 16 + (lane & 15);
    float bv = bias ? bias[n] : 0.f;
#pragma unroll
    for (int r = 0; r < 8; ++r) {
      int m = m0 + r + rowoff;
      float v = acc[j][r] + bv;
      size_t oi = (size_t)m * N + n;
      if (mode == 0)      out32[oi] = v;
      else if (mode == 1) out32[oi] = res[oi] + v;
      else if (mode == 2) out16[oi] = (_Float16)v;
      else if (mode == 3) out16[oi] = (_Float16)gelu_exact(v);
      else {
        int bb = m / vtT, t = m - bb * vtT;
        out16[((size_t)bb * N + n) * vtT + t] = (_Float16)v;
      }
    }
  }
}

// ---------------------------------------------------------------------------
// per-batch mean/rstd over conv output (per = 1024*384)
// ---------------------------------------------------------------------------
__global__ __launch_bounds__(256) void k_batch_stats(const float* __restrict__ c,
                                                     float* __restrict__ stats, int per) {
  int b = blockIdx.x;
  const float* p = c + (size_t)b * per;
  float s = 0.f, ss = 0.f;
  for (int i = threadIdx.x; i < per; i += 256) { float v = p[i]; s += v; ss += v * v; }
  for (int m = 16; m >= 1; m >>= 1) {
    s += __shfl_xor(s, m, 32); ss += __shfl_xor(ss, m, 32);
  }
  __shared__ float sh[16];
  int wave = threadIdx.x >> 5, lane = threadIdx.x & 31;
  if (lane == 0) { sh[wave] = s; sh[8 + wave] = ss; }
  __syncthreads();
  if (threadIdx.x == 0) {
    float S = 0.f, SS = 0.f;
    for (int w = 0; w < 8; ++w) { S += sh[w]; SS += sh[8 + w]; }
    float mu = S / per, var = SS / per - mu * mu;
    stats[b * 2] = mu;
    stats[b * 2 + 1] = rsqrtf(var + 1e-5f);
  }
}

// normalize (per-batch) * ln_g[o][s] + ln_b, exact gelu -> y (B,1024,384) f32
__global__ __launch_bounds__(256) void k_conv_norm_gelu(
    const float* __restrict__ c, const float* __restrict__ stats,
    const float* __restrict__ g, const float* __restrict__ bsh,
    float* __restrict__ y, int n) {
  int i = blockIdx.x * 256 + threadIdx.x;
  if (i >= n) return;
  int b = i / (1024 * 384);
  int rem = i - b * (1024 * 384);
  int s = rem / 384;
  int o = rem - s * 384;
  float mu = stats[b * 2], rstd = stats[b * 2 + 1];
  int gi = o * 1024 + s;                       // ln_g is (E, Hd, Wd)
  float v = (c[i] - mu) * rstd * g[gi] + bsh[gi];
  y[i] = gelu_exact(v);
}

// ---------------------------------------------------------------------------
// token layernorm over E=384: y f32 -> h f16 (one wave per token)
// ---------------------------------------------------------------------------
__global__ __launch_bounds__(256) void k_layernorm(const float* __restrict__ x,
                                                   const float* __restrict__ g,
                                                   const float* __restrict__ b,
                                                   _Float16* __restrict__ out) {
  int wave = threadIdx.x >> 5, lane = threadIdx.x & 31;
  int tok = blockIdx.x * 8 + wave;
  const float* xr = x + (size_t)tok * 384;
  float v[12]; float s = 0.f, ss = 0.f;
#pragma unroll
  for (int j = 0; j < 12; ++j) { v[j] = xr[lane + j * 32]; s += v[j]; ss += v[j] * v[j]; }
  for (int m = 16; m >= 1; m >>= 1) {
    s += __shfl_xor(s, m, 32); ss += __shfl_xor(ss, m, 32);
  }
  float mu = s * (1.f / 384.f);
  float var = ss * (1.f / 384.f) - mu * mu;
  float rstd = rsqrtf(var + 1e-5f);
#pragma unroll
  for (int j = 0; j < 12; ++j) {
    int e = lane + j * 32;
    out[(size_t)tok * 384 + e] = (_Float16)((v[j] - mu) * rstd * g[e] + b[e]);
  }
}

// ---------------------------------------------------------------------------
// qkv f32 (B,S,1152) -> qf/kf (B,4,S,96) f16, vt (B,4,96,S) f16
// ---------------------------------------------------------------------------
__global__ __launch_bounds__(256) void k_split_qkv(const float* __restrict__ qkv,
                                                   _Float16* __restrict__ qf,
                                                   _Float16* __restrict__ kf,
                                                   _Float16* __restrict__ vt, int n) {
  int i = blockIdx.x * 256 + threadIdx.x;
  if (i >= n) return;
  int b = i / (1024 * 1152);
  int rem = i - b * (1024 * 1152);
  int s = rem / 1152;
  int col = rem - s * 1152;
  int w = col / 384;
  int rc = col - w * 384;
  int h = rc / 96;
  int d = rc - h * 96;
  _Float16 v = (_Float16)qkv[i];
  if (w == 0)      qf[((size_t)(b * 4 + h) * 1024 + s) * 96 + d] = v;
  else if (w == 1) kf[((size_t)(b * 4 + h) * 1024 + s) * 96 + d] = v;
  else             vt[((size_t)(b * 4 + h) * 96 + d) * 1024 + s] = v;
}

// ---------------------------------------------------------------------------
// flash attention: grid = B*4heads*16, block = 128 (4 waves x 16 rows)
// qf/kf (B,4,1024,96), vt (B,4,96,1024)  ->  of16 (B,1024,384) merged heads
// K/V tiles double-buffered, staged asynchronously.
// ---------------------------------------------------------------------------
__global__ __launch_bounds__(128) void k_attn(const _Float16* __restrict__ qf,
                                              const _Float16* __restrict__ kf,
                                              const _Float16* __restrict__ vt,
                                              _Float16* __restrict__ of16) {
  __shared__ _Float16 ktile[2][32][96];
  __shared__ _Float16 vtile[2][96][32];
  __shared__ _Float16 pbuf[4][16][32];
  const int lane = threadIdx.x & 31;
  const int wave = threadIdx.x >> 5;
  const int sel  = (lane & 16) ? 8 : 0;
  const int rowoff = (lane & 16) ? 8 : 0;
  const int bh = blockIdx.x >> 4;            // b*4 + h
  const int m0 = (blockIdx.x & 15) * 64 + wave * 16;
  const float rsd = 0.10206207261596577f;    // 96^-0.5

  // preload q fragments: 16 rows x 96 (3 K-chunks of 32)
  const _Float16* qrow = qf + ((size_t)bh * 1024 + m0 + (lane & 15)) * 96;
  v16h aq[3];
#pragma unroll
  for (int kk = 0; kk < 3; ++kk) {
    FragH f;
    f.u[0] = *(const uint4*)(qrow + kk * 32 + sel);
    f.u[1] = *(const uint4*)(qrow + kk * 32 + sel + 16);
    aq[kk] = f.h;
  }

  auto stageKV = [&](int it) {
    int bsel = it & 1;
    int t0 = it * 32;
    // K tile: 32 rows x 96 halves, fully contiguous (6144 B)
    const _Float16* ks = kf + ((size_t)bh * 1024 + t0) * 96 + (size_t)threadIdx.x * 24;
    _Float16* kd = &ktile[bsel][0][0] + (size_t)threadIdx.x * 24;
#pragma unroll
    for (int i = 0; i < 3; ++i) stage_b128(ks + i * 8, kd + i * 8);
    // V tile: 96 rows x 32 halves (64 B per row)
    if (threadIdx.x < 96) {
      const _Float16* vs = vt + ((size_t)bh * 96 + threadIdx.x) * 1024 + t0;
#pragma unroll
      for (int i = 0; i < 4; ++i) stage_b128(vs + i * 8, &vtile[bsel][threadIdx.x][i * 8]);
    }
  };

  v8f o[6];
  for (int j = 0; j < 6; ++j)
    for (int r = 0; r < 8; ++r) o[j][r] = 0.f;
  float mrun[8], lrun[8];
#pragma unroll
  for (int r = 0; r < 8; ++r) { mrun[r] = -1e30f; lrun[r] = 0.f; }

  stageKV(0);
  for (int it = 0; it < 32; ++it) {
    stage_wait();
    __syncthreads();                   // tiles for it landed; it-1 fully consumed
    if (it + 1 < 32) stageKV(it + 1);  // DMA next tiles during compute
    const int bsel = it & 1;

    // scores: two 16x16 C tiles over 32 keys
    v8f s0, s1;
    for (int r = 0; r < 8; ++r) { s0[r] = 0.f; s1[r] = 0.f; }
#pragma unroll
    for (int kk = 0; kk < 3; ++kk) {
      FragH b0, b1;
      const _Float16* k0p = &ktile[bsel][lane & 15][kk * 32 + sel];
      b0.u[0] = *(const uint4*)k0p;
      b0.u[1] = *(const uint4*)(k0p + 16);
      const _Float16* k1p = &ktile[bsel][16 + (lane & 15)][kk * 32 + sel];
      b1.u[0] = *(const uint4*)k1p;
      b1.u[1] = *(const uint4*)(k1p + 16);
      s0 = wmma16(aq[kk], b0.h, s0);
      s1 = wmma16(aq[kk], b1.h, s1);
    }

    // online softmax per row (C layout: row = r + rowoff, 16 lanes per group)
    float p0[8], p1[8], scl[8];
#pragma unroll
    for (int r = 0; r < 8; ++r) {
      float a0 = s0[r] * rsd, a1 = s1[r] * rsd;
      float mx = fmaxf(a0, a1);
      for (int m = 1; m < 16; m <<= 1) mx = fmaxf(mx, __shfl_xor(mx, m, 32));
      float mn = fmaxf(mrun[r], mx);
      scl[r] = __expf(mrun[r] - mn);
      mrun[r] = mn;
      float e0 = __expf(a0 - mn), e1 = __expf(a1 - mn);
      p0[r] = e0; p1[r] = e1;
      float ps = e0 + e1;
      for (int m = 1; m < 16; m <<= 1) ps += __shfl_xor(ps, m, 32);
      lrun[r] = lrun[r] * scl[r] + ps;
    }
#pragma unroll
    for (int j = 0; j < 6; ++j)
      for (int r = 0; r < 8; ++r) o[j][r] *= scl[r];

    // transpose P (C layout -> A layout) through LDS
    {
      int cb = lane & 15;
#pragma unroll
      for (int r = 0; r < 8; ++r) {
        pbuf[wave][r + rowoff][cb]      = (_Float16)p0[r];
        pbuf[wave][r + rowoff][cb + 16] = (_Float16)p1[r];
      }
    }
    __syncthreads();
    FragH pf;
    const _Float16* pp = &pbuf[wave][lane & 15][sel];
    pf.u[0] = *(const uint4*)pp;
    pf.u[1] = *(const uint4*)(pp + 16);
#pragma unroll
    for (int j = 0; j < 6; ++j) {
      FragH bf;
      const _Float16* vp = &vtile[bsel][j * 16 + (lane & 15)][sel];
      bf.u[0] = *(const uint4*)vp;
      bf.u[1] = *(const uint4*)(vp + 16);
      o[j] = wmma16(pf.h, bf.h, o[j]);
    }
  }

  // write merged-head output as f16
  const int b = bh >> 2, h = bh & 3;
#pragma unroll
  for (int j = 0; j < 6; ++j) {
#pragma unroll
    for (int r = 0; r < 8; ++r) {
      int m = m0 + r + rowoff;
      float ov = o[j][r] / lrun[r];
      of16[((size_t)b * 1024 + m) * 384 + h * 96 + j * 16 + (lane & 15)] = (_Float16)ov;
    }
  }
}

// ---------------------------------------------------------------------------
// gaussian head: base = y @ gp_w^T + gp_b, then precompute per-token params
// gparams[tok*8] = {mux, muy, inv_sx, inv_sy, rho, cexp, log_norm, 0}
// ---------------------------------------------------------------------------
__global__ __launch_bounds__(256) void k_gp(const float* __restrict__ y,
                                            const float* __restrict__ gpw,
                                            const float* __restrict__ gpb,
                                            float* __restrict__ gparams) {
  int wave = threadIdx.x >> 5, lane = threadIdx.x & 31;
  int tok = blockIdx.x * 8 + wave;
  const float* yr = y + (size_t)tok * 384;
  float acc[5] = {0.f, 0.f, 0.f, 0.f, 0.f};
  for (int e = lane; e < 384; e += 32) {
    float v = yr[e];
#pragma unroll
    for (int j = 0; j < 5; ++j) acc[j] += v * gpw[j * 384 + e];
  }
#pragma unroll
  for (int j = 0; j < 5; ++j)
    for (int m = 16; m >= 1; m >>= 1) acc[j] += __shfl_xor(acc[j], m, 32);
  if (lane == 0) {
    int s = tok & 1023;               // token within batch (Hd*Wd grid)
    float px = fmaf((float)(s & 31), 2.f / 31.f, -1.f);
    float py = fmaf((float)(s >> 5), 2.f / 31.f, -1.f);
    float b0 = acc[0] + gpb[0], b1 = acc[1] + gpb[1];
    float b2 = acc[2] + gpb[2], b3 = acc[3] + gpb[3], b4 = acc[4] + gpb[4];
    float mux = px + 0.2f * tanhf(b0);
    float muy = py + 0.2f * tanhf(b1);
    float b2c = fminf(fmaxf(b2, -4.f), 4.f);
    float b3c = fminf(fmaxf(b3, -4.f), 4.f);
    float sx = __expf(b2c), sy = __expf(b3c);
    float rho = 0.9f * tanhf(b4);
    float onem = 1.f - rho * rho + 1e-5f;
    float* g = gparams + (size_t)tok * 8;
    g[0] = mux; g[1] = muy;
    g[2] = 1.f / (sx + 1e-5f);
    g[3] = 1.f / (sy + 1e-5f);
    g[4] = rho;
    g[5] = -0.5f / onem;
    g[6] = 1.8378770664093453f + b2c + b3c + 0.5f * logf(onem);  // log(2pi)+...
    g[7] = 0.f;
  }
}

// ---------------------------------------------------------------------------
// final gaussian attention: out[b,g,e] = softmax_t(log_g) @ v
// vt: (B, 384, 4096) f16.  grid = B*64 (16 gaussians per block),
// block = 128: 4 waves, wave w covers e in [w*96, w*96+96).
// scores computed directly in WMMA A-fragment layout; flash softmax.
// ---------------------------------------------------------------------------
__global__ __launch_bounds__(128) void k_final(const float* __restrict__ gparams,
                                               const _Float16* __restrict__ vt,
                                               float* __restrict__ out) {
  const int lane = threadIdx.x & 31;
  const int wave = threadIdx.x >> 5;
  const int sel = (lane & 16) ? 8 : 0;
  const int rowoff = (lane & 16) ? 8 : 0;
  const int b = blockIdx.x >> 6;
  const int g0 = (blockIdx.x & 63) * 16;
  const int e0 = wave * 96;

  const float* gp = gparams + ((size_t)b * 1024 + g0 + (lane & 15)) * 8;
  float mux = gp[0], muy = gp[1], isx = gp[2], isy = gp[3];
  float rho = gp[4], cexp = gp[5], lnorm = gp[6];

  v8f o[6];
  for (int j = 0; j < 6; ++j)
    for (int r = 0; r < 8; ++r) o[j][r] = 0.f;
  float mrun = -1e30f, lrun = 0.f;
  const float gs = 2.f / 63.f;

  for (int t0 = 0; t0 < 4096; t0 += 32) {
    float sc[16];
    float tmax = -1e30f;
#pragma unroll
    for (int i = 0; i < 16; ++i) {
      int koff = (i & 7) + ((i >> 3) << 4) + sel;
      int t = t0 + koff;
      float tx = fmaf((float)(t & 63), gs, -1.f);
      float ty = fmaf((float)(t >> 6), gs, -1.f);
      float zx = (tx - mux) * isx;
      float zy = (ty - muy) * isy;
      float ex = cexp * (zx * zx + zy * zy - 2.f * rho * zx * zy);
      ex = fminf(fmaxf(ex, -70.f), 70.f) - lnorm;
      sc[i] = ex;
      tmax = fmaxf(tmax, ex);
    }
    tmax = fmaxf(tmax, __shfl_xor(tmax, 16, 32));
    float mnew = fmaxf(mrun, tmax);
    float scale = __expf(mrun - mnew);
    mrun = mnew;
    float psum = 0.f;
    FragH pf;
#pragma unroll
    for (int i = 0; i < 16; ++i) {
      float e = __expf(sc[i] - mnew);
      psum += e;
      pf.h[i] = (_Float16)e;
    }
    psum += __shfl_xor(psum, 16, 32);
    lrun = lrun * scale + psum;

    // broadcast per-row scale into C layout and rescale accumulators
    float sr[8];
#pragma unroll
    for (int r = 0; r < 8; ++r) sr[r] = __shfl(scale, r + rowoff, 32);
#pragma unroll
    for (int j = 0; j < 6; ++j)
      for (int r = 0; r < 8; ++r) o[j][r] *= sr[r];

#pragma unroll
    for (int j = 0; j < 6; ++j) {
      const _Float16* vr =
          vt + ((size_t)b * 384 + e0 + j * 16 + (lane & 15)) * 4096 + t0 + sel;
      FragH bf;
      bf.u[0] = *(const uint4*)vr;
      bf.u[1] = *(const uint4*)(vr + 16);
      o[j] = wmma16(pf.h, bf.h, o[j]);
    }
  }

  float linv = 1.f / (lrun + 1e-10f);
  float lr[8];
#pragma unroll
  for (int r = 0; r < 8; ++r) lr[r] = __shfl(linv, r + rowoff, 32);
#pragma unroll
  for (int j = 0; j < 6; ++j)
    for (int r = 0; r < 8; ++r) {
      int g = g0 + r + rowoff;
      out[((size_t)b * 1024 + g) * 384 + e0 + j * 16 + (lane & 15)] = o[j][r] * lr[r];
    }
}

// ---------------------------------------------------------------------------
// host side
// ---------------------------------------------------------------------------
extern "C" void kernel_launch(void* const* d_in, const int* in_sizes, int n_in,
                              void* d_out, int out_size, void* d_ws, size_t ws_size,
                              hipStream_t stream) {
  (void)in_sizes; (void)n_in; (void)out_size; (void)ws_size;
  const int B = 16, S = 1024, E = 384, T = 4096;

  const float* x     = (const float*)d_in[0];
  const float* convw = (const float*)d_in[1];
  const float* convb = (const float*)d_in[2];
  const float* lng   = (const float*)d_in[3];
  const float* lnb   = (const float*)d_in[4];
  const float* blkp[2][12];
  for (int blk = 0; blk < 2; ++blk)
    for (int j = 0; j < 12; ++j) blkp[blk][j] = (const float*)d_in[5 + blk * 12 + j];
  const float* gpw = (const float*)d_in[29];
  const float* gpb = (const float*)d_in[30];
  const float* vw  = (const float*)d_in[31];
  const float* vb  = (const float*)d_in[32];

  // bump allocator on workspace
  char* wp = (char*)d_ws;
  auto alloc = [&](size_t bytes) -> void* {
    void* r = (void*)wp;
    wp += (bytes + 255) & ~(size_t)255;
    return r;
  };
  _Float16* wp16    = (_Float16*)alloc((size_t)384 * 1536 * 2);
  _Float16* qkvw16[2], *projw16[2], *fc1w16[2], *fc2w16[2];
  for (int blk = 0; blk < 2; ++blk) {
    qkvw16[blk] = (_Float16*)alloc((size_t)1152 * 384 * 2);
    projw16[blk] = (_Float16*)alloc((size_t)384 * 384 * 2);
    fc1w16[blk] = (_Float16*)alloc((size_t)1536 * 384 * 2);
    fc2w16[blk] = (_Float16*)alloc((size_t)384 * 1536 * 2);
  }
  _Float16* vw16    = (_Float16*)alloc((size_t)384 * 384 * 2);
  _Float16* xp      = (_Float16*)alloc((size_t)B * S * 1536 * 2);  // reused as hf
  float*    convout = (float*)alloc((size_t)B * S * E * 4);
  float*    stats   = (float*)alloc(256);
  float*    y       = (float*)alloc((size_t)B * S * E * 4);
  _Float16* h16     = (_Float16*)alloc((size_t)B * S * E * 2);
  float*    qkv32   = (float*)alloc((size_t)B * S * 1152 * 4);     // reused as xh
  _Float16* qf      = (_Float16*)alloc((size_t)B * 4 * S * 96 * 2);
  _Float16* kf      = (_Float16*)alloc((size_t)B * 4 * S * 96 * 2);
  _Float16* vta     = (_Float16*)alloc((size_t)B * 4 * 96 * S * 2);
  _Float16* of16    = (_Float16*)alloc((size_t)B * S * E * 2);
  float*    gparams = (float*)alloc((size_t)B * S * 8 * 4);
  _Float16* vtf     = (_Float16*)alloc((size_t)B * E * T * 2);
  _Float16* hf      = xp;                      // alias (xp dead after conv GEMM)
  _Float16* xh      = (_Float16*)qkv32;        // alias (qkv dead after blocks)

  auto cvt = [&](const float* src, _Float16* dst, int n) {
    k_cvt<<<(n + 255) / 256, 256, 0, stream>>>(src, dst, n);
  };
  auto gemm = [&](const _Float16* A, const _Float16* W, const float* bias,
                  int M, int N, int K, int mode, float* o32, const float* res,
                  _Float16* o16, int vtT) {
    dim3 grid(N / 64, M / 128);
    k_gemm<<<grid, 256, 0, stream>>>(A, W, bias, M, N, K, mode, o32, res, o16, vtT);
  };

  // --- weight preparation (f16) ---
  {
    int n = 384 * 1536;
    k_gather_wp<<<(n + 255) / 256, 256, 0, stream>>>(convw, wp16, n);
  }
  for (int blk = 0; blk < 2; ++blk) {
    cvt(blkp[blk][2], qkvw16[blk], 1152 * 384);
    cvt(blkp[blk][4], projw16[blk], 384 * 384);
    cvt(blkp[blk][8], fc1w16[blk], 1536 * 384);
    cvt(blkp[blk][10], fc2w16[blk], 384 * 1536);
  }
  cvt(vw, vw16, 384 * 384);

  // --- patch conv as GEMM ---
  {
    int n = B * S * 1536;
    k_gather_xp<<<(n + 255) / 256, 256, 0, stream>>>(x, xp, n);
  }
  gemm(xp, wp16, convb, B * S, E, 1536, 0, convout, nullptr, nullptr, 0);
  k_batch_stats<<<B, 256, 0, stream>>>(convout, stats, S * E);
  {
    int n = B * S * E;
    k_conv_norm_gelu<<<(n + 255) / 256, 256, 0, stream>>>(convout, stats, lng, lnb, y, n);
  }

  // --- transformer blocks ---
  for (int blk = 0; blk < 2; ++blk) {
    const float* n1g = blkp[blk][0]; const float* n1b = blkp[blk][1];
    const float* qkvb = blkp[blk][3]; const float* projb = blkp[blk][5];
    const float* n2g = blkp[blk][6]; const float* n2b = blkp[blk][7];
    const float* fc1b = blkp[blk][9]; const float* fc2b = blkp[blk][11];

    k_layernorm<<<(B * S) / 8, 256, 0, stream>>>(y, n1g, n1b, h16);
    gemm(h16, qkvw16[blk], qkvb, B * S, 1152, E, 0, qkv32, nullptr, nullptr, 0);
    {
      int n = B * S * 1152;
      k_split_qkv<<<(n + 255) / 256, 256, 0, stream>>>(qkv32, qf, kf, vta, n);
    }
    k_attn<<<B * 4 * 16, 128, 0, stream>>>(qf, kf, vta, of16);
    gemm(of16, projw16[blk], projb, B * S, E, E, 1, y, y, nullptr, 0);

    k_layernorm<<<(B * S) / 8, 256, 0, stream>>>(y, n2g, n2b, h16);
    gemm(h16, fc1w16[blk], fc1b, B * S, 1536, E, 3, nullptr, nullptr, hf, 0);
    gemm(hf, fc2w16[blk], fc2b, B * S, E, 1536, 1, y, y, nullptr, 0);
  }

  // --- gaussian parameters ---
  k_gp<<<(B * S) / 8, 256, 0, stream>>>(y, gpw, gpb, gparams);

  // --- v = x @ v_w^T + v_b, stored transposed per batch: vtf (B, E, T) ---
  cvt(x, xh, B * T * E);
  gemm(xh, vw16, vb, B * T, E, E, 4, nullptr, nullptr, vtf, T);

  // --- final gaussian attention ---
  k_final<<<B * 64, 128, 0, stream>>>(gparams, vtf, (float*)d_out);
}
